// Transformer_17471926960223
// MI455X (gfx1250) — matrix operand order repaired and checked by reading, not compiled, not run on hardware
//
#include <hip/hip_runtime.h>
#include <hip/hip_bf16.h>

typedef unsigned short u16;
typedef __attribute__((ext_vector_type(16))) __bf16 v16bf;
typedef __attribute__((ext_vector_type(8)))  float  v8f;

constexpr int kB = 8, kS = 256, kD = 512, kH = 8, kDK = 256, kHD = 2048, kDFF = 2048, kL = 4;
constexpr int kM = kB * kS;  // 2048 token rows

enum { MODE_QK = 0, MODE_VT = 1, MODE_SCORES = 2, MODE_PV = 3, MODE_RESID = 4, MODE_RELU = 5 };

__device__ __forceinline__ u16 f2bf(float f) {
  unsigned int u = __float_as_uint(f);
  u += 0x7fffu + ((u >> 16) & 1u);   // round-to-nearest-even
  return (u16)(u >> 16);
}

union FragU { uint4 q[2]; v16bf v; };

// ---------------------------------------------------------------------------
// x = data*sqrt(D) + positional encoding + seg_emb[view_idx*S]
// S = 256 = 2^8  ->  S^(2d/D) = 2^(d/32)
// ---------------------------------------------------------------------------
__global__ void embed_kernel(const float* __restrict__ data,
                             const float* __restrict__ seg,
                             const int*   __restrict__ view_idx,
                             float* __restrict__ x) {
  int idx = blockIdx.x * blockDim.x + threadIdx.x;      // over B*S*D
  int d = idx & (kD - 1);
  int s = (idx >> 9) & (kS - 1);
  float ang = (float)s * exp2f(-(float)d * (1.0f / 32.0f));
  float pe = (d & 1) ? cosf(ang) : sinf(ang);
  int srow = view_idx[0] * kS;
  x[idx] = data[idx] * 22.62741699796952f + pe + seg[(size_t)srow * kD + d];
}

// ---------------------------------------------------------------------------
// LayerNorm over D=512, unbiased std (ddof=1), eps added to std.
// One wave32 per row, 16 elements per lane.
// ---------------------------------------------------------------------------
__global__ __launch_bounds__(256) void ln_kernel(const float* __restrict__ x,
                                                 const float* __restrict__ alpha,
                                                 const float* __restrict__ bias,
                                                 u16*   __restrict__ out_bf,
                                                 float* __restrict__ out_f) {
  int row  = blockIdx.x * 8 + (threadIdx.x >> 5);
  int lane = threadIdx.x & 31;
  const float* p = x + (size_t)row * kD;
  float v[16];
  float sum = 0.f;
#pragma unroll
  for (int j = 0; j < 16; ++j) { v[j] = p[lane + 32 * j]; sum += v[j]; }
#pragma unroll
  for (int o = 16; o >= 1; o >>= 1) sum += __shfl_xor(sum, o, 32);
  float mu = sum * (1.0f / 512.0f);
  float ss = 0.f;
#pragma unroll
  for (int j = 0; j < 16; ++j) { float t = v[j] - mu; ss += t * t; }
#pragma unroll
  for (int o = 16; o >= 1; o >>= 1) ss += __shfl_xor(ss, o, 32);
  float inv = 1.0f / (sqrtf(ss * (1.0f / 511.0f)) + 1e-6f);
#pragma unroll
  for (int j = 0; j < 16; ++j) {
    int d = lane + 32 * j;
    float o = alpha[d] * (v[j] - mu) * inv + bias[d];
    if (out_bf) out_bf[(size_t)row * kD + d] = f2bf(o);
    else        out_f [(size_t)row * kD + d] = o;
  }
}

// ---------------------------------------------------------------------------
// fp32 [K,N] -> bf16 transposed [N,K] via LDS tile (coalesced both sides)
// K, N multiples of 32.
// ---------------------------------------------------------------------------
__global__ void transpose_bf16(const float* __restrict__ in, u16* __restrict__ out,
                               int K, int N) {
  __shared__ float tile[32][33];
  int kb = blockIdx.x * 32, nb = blockIdx.y * 32;
  for (int i = threadIdx.y; i < 32; i += 8)
    tile[i][threadIdx.x] = in[(size_t)(kb + i) * N + nb + threadIdx.x];
  __syncthreads();
  for (int i = threadIdx.y; i < 32; i += 8)
    out[(size_t)(nb + i) * K + kb + threadIdx.x] = f2bf(tile[threadIdx.x][i]);
}

// ---------------------------------------------------------------------------
// bf16 WMMA GEMM:  C[M,N] = A[M,K] (row-major) x Bt[N,K]^T  (+ epilogue)
// One wave -> 32x64 output tile: 2 M-frags x 4 N-frags, 8 f32 accumulators.
// All 12 K-step loads issued before the 8-WMMA burst; the compiler software-
// pipelines across K-steps with staggered partial s_wait_loadcnt. Latency is
// hidden by 8 waves/block occupancy (data is L2-resident: <=100MB vs 192MB).
// ---------------------------------------------------------------------------
__global__ __launch_bounds__(256) void gemm_bf16_wmma(
    const u16* __restrict__ A,  long long strideA, int lda,
    const u16* __restrict__ Bt, long long strideB, int ldb,
    int M, int N, int K,
    const float* __restrict__ bias, float scale,
    void* __restrict__ outp, int mode) {
  int lane = threadIdx.x & 31;
  int hf   = lane >> 4;        // half-wave id (0/1)
  int col  = lane & 15;
  int batch = blockIdx.y;
  int tiles_n = N >> 6;
  int gw = blockIdx.x * (blockDim.x >> 5) + (threadIdx.x >> 5);
  int mt = gw / tiles_n, nt = gw - mt * tiles_n;
  int m0 = mt << 5, n0 = nt << 6;          // 32 x 64 tile
  if (m0 >= M) return;

  A  += (long long)batch * strideA;
  Bt += (long long)batch * strideB;

  v8f acc[2][4] = {{v8f{}, v8f{}, v8f{}, v8f{}}, {v8f{}, v8f{}, v8f{}, v8f{}}};
  // A fragment (16x32 bf16): lane m=col; chunks at K offsets 8*hf and 16+8*hf
  const u16* arow0 = A + (size_t)(m0 + col) * lda + 8 * hf;
  const u16* arow1 = arow0 + (size_t)16 * lda;
  // B fragment (32x16 bf16): lane n=col; chunks at K offsets 8*hf and 16+8*hf
  const u16* brow = Bt + (size_t)(n0 + col) * ldb + 8 * hf;

  for (int k0 = 0; k0 < K; k0 += 32) {
    FragU a0, a1, b[4];
    a0.q[0] = *(const uint4*)(arow0 + k0);
    a0.q[1] = *(const uint4*)(arow0 + k0 + 16);
    a1.q[0] = *(const uint4*)(arow1 + k0);
    a1.q[1] = *(const uint4*)(arow1 + k0 + 16);
#pragma unroll
    for (int t = 0; t < 4; ++t) {
      const u16* br = brow + (size_t)(16 * t) * ldb + k0;
      b[t].q[0] = *(const uint4*)(br);
      b[t].q[1] = *(const uint4*)(br + 16);
    }
#pragma unroll
    for (int t = 0; t < 4; ++t) {
      acc[0][t] = __builtin_amdgcn_wmma_f32_16x16x32_bf16(
          false, a0.v, false, b[t].v, (short)0, acc[0][t], false, false);
      acc[1][t] = __builtin_amdgcn_wmma_f32_16x16x32_bf16(
          false, a1.v, false, b[t].v, (short)0, acc[1][t], false, false);
    }
  }

  // C/D layout: VGPR r, lanes 0-15 -> (M=base+r, N), lanes 16-31 -> (M=base+r+8, N)
#pragma unroll
  for (int mi = 0; mi < 2; ++mi) {
#pragma unroll
    for (int t = 0; t < 4; ++t) {
      int n = n0 + 16 * t + col;
      float bvl = bias ? bias[n] : 0.f;
#pragma unroll
      for (int r = 0; r < 8; ++r) {
        int m = m0 + 16 * mi + r + 8 * hf;
        float val = acc[mi][t][r] * scale + bvl;
        if (mode == MODE_QK) {                       // -> [B,H,S,DK] bf16
          int b = m >> 8, s = m & 255, h = n >> 8, d = n & 255;
          ((u16*)outp)[(((size_t)b * kH + h) * kS + s) * kDK + d] = f2bf(val);
        } else if (mode == MODE_VT) {                // -> [B,H,DK,S] bf16
          int b = m >> 8, s = m & 255, h = n >> 8, d = n & 255;
          ((u16*)outp)[(((size_t)b * kH + h) * kDK + d) * kS + s] = f2bf(val);
        } else if (mode == MODE_SCORES) {            // -> fp32 [bh,S,S], scaled
          ((float*)outp)[(size_t)batch * kS * kS + (size_t)m * kS + n] = val;
        } else if (mode == MODE_PV) {                // -> concat heads [B*S, HD] bf16
          int b = batch >> 3, h = batch & 7;
          ((u16*)outp)[(size_t)(b * kS + m) * kHD + h * kDK + n] = f2bf(val);
        } else if (mode == MODE_RESID) {             // x += proj + bias   (fp32)
          ((float*)outp)[(size_t)m * N + n] += val;
        } else {                                     // ReLU -> bf16 [M,N]
          ((u16*)outp)[(size_t)m * N + n] = f2bf(fmaxf(val, 0.f));
        }
      }
    }
  }
}

// ---------------------------------------------------------------------------
// Masked softmax over key axis (256), one wave per (b,h,q) row. bf16 probs out.
// ---------------------------------------------------------------------------
__global__ __launch_bounds__(256) void softmax_kernel(const float* __restrict__ scores,
                                                      const int* __restrict__ attn_m,
                                                      u16* __restrict__ probs) {
  int row  = blockIdx.x * 8 + (threadIdx.x >> 5);   // over B*H*S
  int lane = threadIdx.x & 31;
  int b = row / (kH * kS);
  const float* src  = scores + (size_t)row * kS;
  const int*   mrow = attn_m + (size_t)b * kS;
  float v[8];
  float mx = -3.4e38f;
#pragma unroll
  for (int j = 0; j < 8; ++j) {
    int key = lane + 32 * j;
    float s = src[key];
    if (mrow[key] != 0) s = -1e9f;
    v[j] = s;
    mx = fmaxf(mx, s);
  }
#pragma unroll
  for (int o = 16; o >= 1; o >>= 1) mx = fmaxf(mx, __shfl_xor(mx, o, 32));
  float sum = 0.f;
#pragma unroll
  for (int j = 0; j < 8; ++j) { v[j] = __expf(v[j] - mx); sum += v[j]; }
#pragma unroll
  for (int o = 16; o >= 1; o >>= 1) sum += __shfl_xor(sum, o, 32);
  float inv = 1.0f / sum;
  u16* dst = probs + (size_t)row * kS;
#pragma unroll
  for (int j = 0; j < 8; ++j) dst[lane + 32 * j] = f2bf(v[j] * inv);
}

// ---------------------------------------------------------------------------
extern "C" void kernel_launch(void* const* d_in, const int* in_sizes, int n_in,
                              void* d_out, int out_size, void* d_ws, size_t ws_size,
                              hipStream_t stream) {
  (void)in_sizes; (void)n_in; (void)out_size; (void)ws_size;
  const float* data     = (const float*)d_in[0];
  const int*   attn_m   = (const int*)  d_in[1];
  const int*   view_idx = (const int*)  d_in[2];
  const float* seg_emb  = (const float*)d_in[3];
  const float* Wq = (const float*)d_in[4];   const float* bq = (const float*)d_in[5];
  const float* Wk = (const float*)d_in[6];   const float* bk = (const float*)d_in[7];
  const float* Wv = (const float*)d_in[8];   const float* bv = (const float*)d_in[9];
  const float* Wo = (const float*)d_in[10];  const float* bo = (const float*)d_in[11];
  const float* W1 = (const float*)d_in[12];  const float* b1 = (const float*)d_in[13];
  const float* W2 = (const float*)d_in[14];  const float* b2 = (const float*)d_in[15];
  const float* alpha1 = (const float*)d_in[16]; const float* bias1 = (const float*)d_in[17];
  const float* alpha2 = (const float*)d_in[18]; const float* bias2 = (const float*)d_in[19];
  const float* alpha_f = (const float*)d_in[20]; const float* bias_f = (const float*)d_in[21];

  char* ws = (char*)d_ws;
  size_t off = 0;
  auto alloc = [&](size_t bytes) -> void* {
    void* p = ws + off;
    off += (bytes + 255) & ~(size_t)255;
    return p;
  };
  float* x   = (float*)alloc((size_t)kM * kD * 4);                 // running residual
  u16*   x2  = (u16*)  alloc((size_t)kM * kD * 2);                 // LN output, bf16
  u16*   q   = (u16*)  alloc((size_t)kB * kH * kS * kDK * 2);      // [B,H,S,DK]
  u16*   kk  = (u16*)  alloc((size_t)kB * kH * kS * kDK * 2);      // [B,H,S,DK]
  u16*   vT  = (u16*)  alloc((size_t)kB * kH * kDK * kS * 2);      // [B,H,DK,S]
  float* sc  = (float*)alloc((size_t)kB * kH * kS * kS * 4);       // scores
  u16*   pr  = (u16*)  alloc((size_t)kB * kH * kS * kS * 2);       // probs bf16
  u16*   oc  = (u16*)  alloc((size_t)kM * kHD * 2);                // concat heads
  u16*   fh  = (u16*)  alloc((size_t)kM * kDFF * 2);               // FFN hidden
  u16*   wqT = (u16*)  alloc((size_t)kD * kHD * 2);
  u16*   wkT = (u16*)  alloc((size_t)kD * kHD * 2);
  u16*   wvT = (u16*)  alloc((size_t)kD * kHD * 2);
  u16*   woT = (u16*)  alloc((size_t)kHD * kD * 2);
  u16*   w1T = (u16*)  alloc((size_t)kD * kDFF * 2);
  u16*   w2T = (u16*)  alloc((size_t)kDFF * kD * 2);

  embed_kernel<<<(kM * kD) / 256, 256, 0, stream>>>(data, seg_emb, view_idx, x);

  dim3 tb(32, 8);
  for (int i = 0; i < kL; ++i) {
    transpose_bf16<<<dim3(kD / 32,   kHD / 32), tb, 0, stream>>>(Wq + (size_t)i * kD * kHD, wqT, kD,   kHD);
    transpose_bf16<<<dim3(kD / 32,   kHD / 32), tb, 0, stream>>>(Wk + (size_t)i * kD * kHD, wkT, kD,   kHD);
    transpose_bf16<<<dim3(kD / 32,   kHD / 32), tb, 0, stream>>>(Wv + (size_t)i * kD * kHD, wvT, kD,   kHD);
    transpose_bf16<<<dim3(kHD / 32,  kD / 32),  tb, 0, stream>>>(Wo + (size_t)i * kHD * kD, woT, kHD,  kD);
    transpose_bf16<<<dim3(kD / 32,   kDFF / 32),tb, 0, stream>>>(W1 + (size_t)i * kD * kDFF, w1T, kD,  kDFF);
    transpose_bf16<<<dim3(kDFF / 32, kD / 32),  tb, 0, stream>>>(W2 + (size_t)i * kDFF * kD, w2T, kDFF, kD);

    // ---- sublayer 1: attention ----
    ln_kernel<<<kM / 8, 256, 0, stream>>>(x, alpha1 + i * kD, bias1 + i * kD, x2, nullptr);

    // 32x64 tiles: (2048/32)*(2048/64)=2048 waves -> 256 blocks
    gemm_bf16_wmma<<<dim3(256, 1), 256, 0, stream>>>(x2, 0, kD, wqT, 0, kD,
        kM, kHD, kD, bq + i * kHD, 1.f, q,  MODE_QK);
    gemm_bf16_wmma<<<dim3(256, 1), 256, 0, stream>>>(x2, 0, kD, wkT, 0, kD,
        kM, kHD, kD, bk + i * kHD, 1.f, kk, MODE_QK);
    gemm_bf16_wmma<<<dim3(256, 1), 256, 0, stream>>>(x2, 0, kD, wvT, 0, kD,
        kM, kHD, kD, bv + i * kHD, 1.f, vT, MODE_VT);

    // scores = Q K^T / sqrt(DK): (256/32)*(256/64)=32 waves/batch -> 4 blocks
    gemm_bf16_wmma<<<dim3(4, kB * kH), 256, 0, stream>>>(q, (long long)kS * kDK, kDK,
        kk, (long long)kS * kDK, kDK, kS, kS, kDK, nullptr, 0.0625f, sc, MODE_SCORES);

    softmax_kernel<<<(kB * kH * kS) / 8, 256, 0, stream>>>(sc, attn_m, pr);

    // O = P V, batched; scatter into concat-head layout
    gemm_bf16_wmma<<<dim3(4, kB * kH), 256, 0, stream>>>(pr, (long long)kS * kS, kS,
        vT, (long long)kDK * kS, kS, kS, kDK, kS, nullptr, 1.f, oc, MODE_PV);

    // x += O Wo + bo : (2048/32)*(512/64)=512 waves -> 64 blocks
    gemm_bf16_wmma<<<dim3(64, 1), 256, 0, stream>>>(oc, 0, kHD, woT, 0, kHD,
        kM, kD, kHD, bo + i * kD, 1.f, x, MODE_RESID);

    // ---- sublayer 2: FFN ----
    ln_kernel<<<kM / 8, 256, 0, stream>>>(x, alpha2 + i * kD, bias2 + i * kD, x2, nullptr);

    gemm_bf16_wmma<<<dim3(256, 1), 256, 0, stream>>>(x2, 0, kD, w1T, 0, kD,
        kM, kDFF, kD, b1 + i * kDFF, 1.f, fh, MODE_RELU);
    gemm_bf16_wmma<<<dim3(64, 1), 256, 0, stream>>>(fh, 0, kDFF, w2T, 0, kDFF,
        kM, kD, kDFF, b2 + i * kD, 1.f, x, MODE_RESID);
  }

  ln_kernel<<<kM / 8, 256, 0, stream>>>(x, alpha_f, bias_f, nullptr, (float*)d_out);
}
